// MIGCL_88613765251351
// MI455X (gfx1250) — matrix operand order repaired and checked by reading, not compiled, not run on hardware
//
#include <hip/hip_runtime.h>
#include <math.h>

#define NN   50000
#define EDGES 800000
#define DIN  3000
#define H0D  512
#define H1D  128
#define GHD  256
#define EMBD 128
#define D1D  512

typedef __attribute__((ext_vector_type(16))) __bf16          bf16x16;
typedef __attribute__((ext_vector_type(8)))  float           f32x8;
typedef __attribute__((ext_vector_type(8)))  unsigned short  u16x8;

union FragU { bf16x16 bf; u16x8 u[2]; };

__device__ __forceinline__ unsigned short f2bf(float f) {
  unsigned int u = __float_as_uint(f);
  u += 0x7FFFu + ((u >> 16) & 1u);          // round-to-nearest-even
  return (unsigned short)(u >> 16);
}

__device__ __forceinline__ int iminx(int a, int b) { return a < b ? a : b; }

// ---------------------------------------------------------------------------
// Fused GEMM: out = act( (A@B + bias) * (g*rsqrt(1.001)) + bt )
// A: [M,K] fp32 (Af32) or bf16 (Abf), staged through LDS (software pipelined).
// B: pre-transposed bf16, K-major Bt[N][Kp], Kp = K rounded up to 32 and
//    zero-padded -> WMMA B fragments are 16B-contiguous per lane and are
//    loaded DIRECTLY from global (L2-resident weights), no LDS staging.
// Block: 256 thr = 8 wave32 -> 64M x 128N tile; wave = 16M x 64N via four
// v_wmma_f32_16x16x32_bf16 accumulators; K step 32.
// ---------------------------------------------------------------------------
__global__ __launch_bounds__(256, 2)
void gemm_bf16_wmma(const float* __restrict__ Af32,
                    const unsigned short* __restrict__ Abf,
                    const unsigned short* __restrict__ Bt,   // [N][Kp] bf16
                    int M, int N, int K, int Kp,
                    const float* __restrict__ bias,
                    const float* __restrict__ bnG,
                    const float* __restrict__ bnB,
                    int act,                       // 0 none, 1 elu, 2 sigmoid
                    float* __restrict__ outF,
                    unsigned short* __restrict__ outBf) {
  __shared__ unsigned short As[64 * 40];          // [row][k], pad stride 40

  const int tid  = threadIdx.x;
  const int wave = tid >> 5, lane = tid & 31;
  const int mw = wave & 3, nw = wave >> 2;
  const int half = lane >> 4, l16 = lane & 15;
  const int rowBase = blockIdx.y * 64;
  const int colBase = blockIdx.x * 128;

  // A staging coordinates (fixed per thread): 8 contiguous K elems
  const int ar = tid >> 2;                        // A row in tile   0..63
  const int ak = (tid & 3) * 8;                   // A k offset      0,8,16,24
  const size_t aOff = (size_t)(rowBase + ar) * K + ak;

  // B fragment base pointers (loop-invariant, clamped for edge-N tiles)
  const unsigned short* bptr[4];
  #pragma unroll
  for (int c = 0; c < 4; ++c) {
    const int col = iminx(colBase + nw * 64 + c * 16 + l16, N - 1);
    bptr[c] = Bt + (size_t)col * Kp + half * 16;
  }

  f32x8 acc[4];
  #pragma unroll
  for (int c = 0; c < 4; ++c)
    #pragma unroll
    for (int j = 0; j < 8; ++j) acc[c][j] = 0.0f;

  // pending A tile (pipelined; fp32 converted at store time)
  float4 a0, a1;
  u16x8  aU;
  auto loadA = [&](int k0) {
    if (Af32) {
      const float4* pa = (const float4*)(Af32 + aOff + k0);
      a0 = pa[0]; a1 = pa[1];
    } else {
      aU = *(const u16x8*)(Abf + aOff + k0);
    }
  };
  auto storeA = [&]() {
    u16x8 v;
    if (Af32) {
      v[0] = f2bf(a0.x); v[1] = f2bf(a0.y); v[2] = f2bf(a0.z); v[3] = f2bf(a0.w);
      v[4] = f2bf(a1.x); v[5] = f2bf(a1.y); v[6] = f2bf(a1.z); v[7] = f2bf(a1.w);
    } else {
      v = aU;
    }
    *(u16x8*)&As[ar * 40 + ak] = v;
  };

  // ---- one WMMA step: A frags from LDS, B frags direct from L2 ----
  auto mma = [&](int k0) {
    FragU a, b[4];
    const int arow = mw * 16 + l16;
    // A 16x32 bf16: row = lane%16; elem e -> K = half*16 + e (K-contig)
    a.u[0] = *(const u16x8*)&As[arow * 40 + half * 16];
    a.u[1] = *(const u16x8*)&As[arow * 40 + half * 16 + 8];
    #pragma unroll
    for (int c = 0; c < 4; ++c) {
      // B 32x16 bf16: col = lane%16; elem e -> K = half*16 + e (K-contig)
      b[c].u[0] = *(const u16x8*)(bptr[c] + k0);
      b[c].u[1] = *(const u16x8*)(bptr[c] + k0 + 8);
    }
    #pragma unroll
    for (int c = 0; c < 4; ++c)
      acc[c] = __builtin_amdgcn_wmma_f32_16x16x32_bf16(
          false, a.bf, false, b[c].bf, (short)0, acc[c], false, false);
  };

  // ---- branchless guarded A staging (edge tiles / K tail) ----
  auto stage_guardA = [&](int k0) {
    const int grow = rowBase + ar;
    const int rowC = iminx(grow, M - 1);
    unsigned short* dstA = &As[ar * 40 + ak];
    #pragma unroll
    for (int i = 0; i < 8; ++i) {
      const int kk = k0 + ak + i;
      const int kC = iminx(kk, K - 1);
      unsigned short v;
      if (Af32) v = f2bf(Af32[(size_t)rowC * K + kC]);
      else      v = Abf[(size_t)rowC * K + kC];
      dstA[i] = (grow < M && kk < K) ? v : (unsigned short)0;
    }
  };

  const bool interior = (rowBase + 64 <= M);
  const int kFull = K & ~31;

  if (interior) {
    loadA(0);
    for (int k0 = 0; k0 < kFull; k0 += 32) {
      storeA();
      __syncthreads();
      if (k0 + 32 < kFull) {
        loadA(k0 + 32);                            // overlaps with WMMA below
        if (k0 + 64 < kFull) {                     // global_prefetch_b8 path
          if (Af32) __builtin_prefetch(Af32 + aOff + k0 + 64, 0, 0);
          else      __builtin_prefetch(Abf + aOff + k0 + 64, 0, 0);
        }
      }
      mma(k0);
      __syncthreads();
    }
    if (kFull < K) {                               // K tail (K=3000 only)
      stage_guardA(kFull);
      __syncthreads();
      mma(kFull);                                  // B pad reads zeros
    }
  } else {                                         // edge-M tiles: rare
    for (int k0 = 0; k0 < K; k0 += 32) {
      __syncthreads();
      stage_guardA(k0);
      __syncthreads();
      mma(k0);
    }
  }

  // ---- epilogue: C/D layout row = half*8 + v, col = lane%16 ----
  const float bnRs = rsqrtf(1.001f);
  #pragma unroll
  for (int c = 0; c < 4; ++c) {
    const int col = colBase + nw * 64 + c * 16 + l16;
    if (col >= N) continue;
    const float bi = bias ? bias[col] : 0.0f;
    const float sc = bnG ? bnG[col] * bnRs : 1.0f;
    const float sh = bnB ? bnB[col] : 0.0f;
    #pragma unroll
    for (int v = 0; v < 8; ++v) {
      const int row = rowBase + mw * 16 + half * 8 + v;
      if (row >= M) continue;
      float x = (acc[c][v] + bi) * sc + sh;
      if (act == 1)      x = x > 0.0f ? x : (__expf(x) - 1.0f);   // ELU
      else if (act == 2) x = 1.0f / (1.0f + __expf(-x));          // sigmoid
      const size_t o = (size_t)row * N + col;
      if (outF)  outF[o]  = x;
      if (outBf) outBf[o] = f2bf(x);
    }
  }
}

// ---------------------------------------------------------------------------
// Weight pre-transpose: Bt[n*Kp + k] = bf16(B[k*N + n]), zero-padded K->Kp
// ---------------------------------------------------------------------------
__global__ void transpose_bf_kernel(const float* __restrict__ B,
                                    unsigned short* __restrict__ Bt,
                                    int K, int N, int Kp) {
  const int k = blockIdx.x * blockDim.x + threadIdx.x;
  const int n = blockIdx.y;
  if (k < Kp)
    Bt[(size_t)n * Kp + k] = (k < K) ? f2bf(B[(size_t)k * N + n])
                                     : (unsigned short)0;
}

// ---------------------------------------------------------------------------
// Elementwise helpers
// ---------------------------------------------------------------------------
__global__ void rowbias_init_kernel(float* __restrict__ out,
                                    const float* __restrict__ b,
                                    int cMask, long long n) {
  long long i = (long long)blockIdx.x * blockDim.x + threadIdx.x;
  if (i < n) out[i] = b[(int)(i & cMask)];
}

__global__ void relu_cvt_kernel(const float* __restrict__ in,
                                unsigned short* __restrict__ out, long long n) {
  long long i = (long long)blockIdx.x * blockDim.x + threadIdx.x;
  if (i < n) out[i] = f2bf(fmaxf(in[i], 0.0f));
}

// agg[dst] += w * s[src] ; one thread per edge x 4-column quad (C = 1<<cShift)
__global__ void spmm_scatter4_kernel(const int* __restrict__ srcIdx,
                                     const int* __restrict__ dstIdx,
                                     const float* __restrict__ w,
                                     const float* __restrict__ s,
                                     float* __restrict__ out,
                                     int cShift, int qMask, long long n) {
  long long i = (long long)blockIdx.x * blockDim.x + threadIdx.x;
  if (i >= n) return;
  const int e = (int)(i >> (cShift - 2));
  const int c = (int)(i & qMask) << 2;
  const float we = w[e];
  const float4 v = *(const float4*)&s[((size_t)srcIdx[e] << cShift) + c];
  float* o = &out[((size_t)dstIdx[e] << cShift) + c];
  atomicAdd(o + 0, we * v.x);
  atomicAdd(o + 1, we * v.y);
  atomicAdd(o + 2, we * v.z);
  atomicAdd(o + 3, we * v.w);
}

// softmax over 2 views per node; one wave32 per node, 4 cols per lane
__global__ __launch_bounds__(256)
void attention_kernel(const float* __restrict__ femb,
                      const float* __restrict__ semb,
                      const float* __restrict__ aW,
                      float* __restrict__ embF,
                      unsigned short* __restrict__ embBf, int nNodes) {
  const int wave = threadIdx.x >> 5, lane = threadIdx.x & 31;
  const int n = blockIdx.x * 8 + wave;
  if (n >= nNodes) return;
  const float* f = femb + (size_t)n * EMBD;
  const float* s = semb + (size_t)n * EMBD;
  float pf = 0.0f, ps = 0.0f;
  #pragma unroll
  for (int j = 0; j < 4; ++j) {
    const int c = lane + 32 * j;
    const float av = aW[c];
    pf += f[c] * av;
    ps += s[c] * av;
  }
  #pragma unroll
  for (int m = 16; m >= 1; m >>= 1) {
    pf += __shfl_xor(pf, m, 32);
    ps += __shfl_xor(ps, m, 32);
  }
  const float mx = fmaxf(pf, ps);
  const float e1 = __expf(pf - mx), e2 = __expf(ps - mx);
  const float b1 = e1 / (e1 + e2), b2 = 1.0f - b1;
  #pragma unroll
  for (int j = 0; j < 4; ++j) {
    const int c = lane + 32 * j;
    const float v = b1 * f[c] + b2 * s[c];
    embF[(size_t)n * EMBD + c]  = v;
    embBf[(size_t)n * EMBD + c] = f2bf(v);
  }
}

// ---------------------------------------------------------------------------
extern "C" void kernel_launch(void* const* d_in, const int* in_sizes, int n_in,
                              void* d_out, int out_size, void* d_ws, size_t ws_size,
                              hipStream_t stream) {
  (void)in_sizes; (void)n_in; (void)out_size; (void)ws_size;
  const float* feat = (const float*)d_in[0];
  const int*   fidx = (const int*)d_in[1];
  const float* fw   = (const float*)d_in[2];
  const int*   sidx = (const int*)d_in[3];
  const float* sw   = (const float*)d_in[4];
  const float* eW0  = (const float*)d_in[5];
  const float* eb0  = (const float*)d_in[6];
  const float* eg0  = (const float*)d_in[7];
  const float* ebt0 = (const float*)d_in[8];
  const float* eW1  = (const float*)d_in[9];
  const float* eb1  = (const float*)d_in[10];
  const float* eg1  = (const float*)d_in[11];
  const float* ebt1 = (const float*)d_in[12];
  const float* gW1  = (const float*)d_in[13];
  const float* gb1  = (const float*)d_in[14];
  const float* gW2  = (const float*)d_in[15];
  const float* gb2  = (const float*)d_in[16];
  const float* aW   = (const float*)d_in[17];
  const float* dW0  = (const float*)d_in[18];
  const float* db0  = (const float*)d_in[19];
  const float* dg0  = (const float*)d_in[20];
  const float* dbt0 = (const float*)d_in[21];
  const float* dW1  = (const float*)d_in[22];
  const float* db1  = (const float*)d_in[23];
  const float* dg1  = (const float*)d_in[24];
  const float* dbt1 = (const float*)d_in[25];

  // outputs: (femb, semb, de, emb) concatenated, fp32
  float* out  = (float*)d_out;
  float* femb = out;
  float* semb = out + (size_t)NN * EMBD;
  float* de   = out + (size_t)2 * NN * EMBD;
  float* emb  = out + (size_t)2 * NN * EMBD + (size_t)NN * DIN;

  // padded K strides (round up to 32)
  const int KpDIN = (DIN + 31) & ~31;   // 3008
  const int KpH0  = H0D;                // 512
  const int KpH1  = H1D;                // 128
  const int KpGH  = GHD;                // 256
  const int KpEMB = EMBD;               // 128
  const int KpD1  = D1D;                // 512

  // workspace carve-out (256B aligned)
  char* ws = (char*)d_ws;
  size_t off = 0;
  auto alloc = [&](size_t bytes) -> char* {
    char* p = ws + off;
    off = (off + bytes + 255) & ~(size_t)255;
    return p;
  };
  unsigned short* eW0t = (unsigned short*)alloc((size_t)H0D * KpDIN * 2);
  unsigned short* eW1t = (unsigned short*)alloc((size_t)H1D * KpH0 * 2);
  unsigned short* gW1t = (unsigned short*)alloc((size_t)GHD * KpH1 * 2);
  unsigned short* gW2t = (unsigned short*)alloc((size_t)EMBD * KpGH * 2);
  unsigned short* dW0t = (unsigned short*)alloc((size_t)D1D * KpEMB * 2);
  unsigned short* dW1t = (unsigned short*)alloc((size_t)DIN * KpD1 * 2);
  unsigned short* z0b  = (unsigned short*)alloc((size_t)NN * H0D * 2);
  unsigned short* z1b  = (unsigned short*)alloc((size_t)NN * H1D * 2);
  float*          sbuf = (float*)alloc((size_t)NN * GHD * 4);
  float*          agg1 = (float*)alloc((size_t)NN * GHD * 4);
  unsigned short* hb   = (unsigned short*)alloc((size_t)NN * GHD * 2);
  float*          h2   = (float*)alloc((size_t)NN * EMBD * 4);
  unsigned short* embb = (unsigned short*)alloc((size_t)NN * EMBD * 2);
  unsigned short* d0b  = (unsigned short*)alloc((size_t)NN * D1D * 2);

  auto xpose = [&](const float* B, unsigned short* Bt, int K, int N, int Kp) {
    dim3 grid((unsigned)((Kp + 255) / 256), (unsigned)N);
    transpose_bf_kernel<<<grid, 256, 0, stream>>>(B, Bt, K, N, Kp);
  };
  xpose(eW0, eW0t, DIN, H0D, KpDIN);
  xpose(eW1, eW1t, H0D, H1D, KpH0);
  xpose(gW1, gW1t, H1D, GHD, KpH1);
  xpose(gW2, gW2t, GHD, EMBD, KpGH);
  xpose(dW0, dW0t, EMBD, D1D, KpEMB);
  xpose(dW1, dW1t, D1D, DIN, KpD1);

  auto gemm = [&](const float* Af, const unsigned short* Ab,
                  const unsigned short* Bt, int M, int N, int K, int Kp,
                  const float* bi, const float* g, const float* bt,
                  int act, float* oF, unsigned short* oB) {
    dim3 grid((unsigned)((N + 127) / 128), (unsigned)((M + 63) / 64));
    gemm_bf16_wmma<<<grid, 256, 0, stream>>>(Af, Ab, Bt, M, N, K, Kp,
                                             bi, g, bt, act, oF, oB);
  };

  // encoder: elu(bn(feat@eW0+eb0)) -> elu(bn(z@eW1+eb1))
  gemm(feat, nullptr, eW0t, NN, H0D, DIN, KpDIN, eb0, eg0, ebt0, 1, nullptr, z0b);
  gemm(nullptr, z0b, eW1t, NN, H1D, H0D, KpH0, eb1, eg1, ebt1, 1, nullptr, z1b);
  // shared GCN layer-1 dense part: s = z1 @ gW1
  gemm(nullptr, z1b, gW1t, NN, GHD, H1D, KpH1, nullptr, nullptr, nullptr, 0,
       sbuf, nullptr);

  auto graph = [&](const int* idx, const float* w, float* outEmb) {
    const long long n1 = (long long)NN * GHD;
    rowbias_init_kernel<<<(unsigned)((n1 + 255) / 256), 256, 0, stream>>>(
        agg1, gb1, GHD - 1, n1);
    const long long sc1 = (long long)EDGES * (GHD / 4);
    spmm_scatter4_kernel<<<(unsigned)((sc1 + 255) / 256), 256, 0, stream>>>(
        idx, idx + EDGES, w, sbuf, agg1, 8, (GHD / 4) - 1, sc1);
    relu_cvt_kernel<<<(unsigned)((n1 + 255) / 256), 256, 0, stream>>>(agg1, hb, n1);
    gemm(nullptr, hb, gW2t, NN, EMBD, GHD, KpGH, nullptr, nullptr, nullptr, 0,
         h2, nullptr);
    const long long n2 = (long long)NN * EMBD;
    rowbias_init_kernel<<<(unsigned)((n2 + 255) / 256), 256, 0, stream>>>(
        outEmb, gb2, EMBD - 1, n2);
    const long long sc2 = (long long)EDGES * (EMBD / 4);
    spmm_scatter4_kernel<<<(unsigned)((sc2 + 255) / 256), 256, 0, stream>>>(
        idx, idx + EDGES, w, h2, outEmb, 7, (EMBD / 4) - 1, sc2);
  };
  graph(fidx, fw, femb);
  graph(sidx, sw, semb);

  attention_kernel<<<(unsigned)((NN + 7) / 8), 256, 0, stream>>>(
      femb, semb, aW, emb, embb, NN);

  // decoder: elu(bn(emb@dW0+db0)) -> sigmoid(bn(d@dW1+db1))
  gemm(nullptr, embb, dW0t, NN, D1D, EMBD, KpEMB, db0, dg0, dbt0, 1,
       nullptr, d0b);
  gemm(nullptr, d0b, dW1t, NN, DIN, D1D, KpD1, db1, dg1, dbt1, 2, de, nullptr);
}
